// JVPBlockCirculant_34565896798393
// MI455X (gfx1250) — matrix operand order repaired and checked by reading, not compiled, not run on hardware
//
#include <hip/hip_runtime.h>

// Block-circulant matmul y = x * B^T + bias on gfx1250 via bf16 hi/lo-split WMMA.
// B[i*128+t, j*128+u] = W[i,j,(u-t) mod 128].  x:(8192,4096) f32, W:(32,32,128) f32.

typedef __attribute__((ext_vector_type(16))) __bf16 v16bf;
typedef __attribute__((ext_vector_type(8)))  float  v8f;

union BFrag { v16bf v; unsigned int u[8]; };

static constexpr int DIN     = 4096;
static constexpr int DOUT    = 4096;
static constexpr int CB      = 128;   // circulant block size
static constexpr int TILE_M  = 128;
static constexpr int TILE_K  = 32;
static constexpr int NTHR    = 256;   // 8 wave32s

__device__ __forceinline__ void bf16_split(float f, unsigned int& hi, unsigned int& lo) {
  // round-to-nearest-even f32 -> bf16, then residual -> bf16
  unsigned int u  = __float_as_uint(f);
  unsigned int r  = u + 0x7FFFu + ((u >> 16) & 1u);
  hi = r >> 16;
  float hf = __uint_as_float(r & 0xFFFF0000u);
  float lf = f - hf;
  unsigned int ul = __float_as_uint(lf);
  unsigned int r2 = ul + 0x7FFFu + ((ul >> 16) & 1u);
  lo = r2 >> 16;
}

__global__ __launch_bounds__(NTHR)
void bc_wmma_kernel(const float* __restrict__ x,
                    const float* __restrict__ W,
                    const float* __restrict__ dvec,
                    const float* __restrict__ bias,
                    float* __restrict__ y)
{
  // W[iblk] staged as bf16, circulant-duplicated to length 256 per row (j),
  // plus a copy shifted by one element so odd rotations read aligned u32 pairs.
  __shared__ __align__(16) unsigned short s_whi [32 * 256];
  __shared__ __align__(16) unsigned short s_whiS[32 * 256];
  __shared__ __align__(16) unsigned short s_wlo [32 * 256];
  __shared__ __align__(16) unsigned short s_wloS[32 * 256];
  // x tile 128 rows x 32 K, as packed bf16 pairs (u32), stride 18 u32 (bank pad)
  __shared__ __align__(16) unsigned int   s_xhi [128 * 18];
  __shared__ __align__(16) unsigned int   s_xlo [128 * 18];

  const int tid  = threadIdx.x;
  const int lane = tid & 31;
  const int wave = tid >> 5;
  const int wm   = wave & 3;      // 4 waves along M
  const int wn   = wave >> 2;     // 2 waves along N
  const int lg   = lane >> 4;     // lane group 0/1
  const int ln   = lane & 15;

  const int m0   = blockIdx.x * TILE_M;
  const int iblk = blockIdx.y;
  const int n0   = iblk * CB;

  // ---- stage W[iblk] (32x128 f32 = 16KB) into the 4 LDS arrays ----
  {
    const float* Wsl = W + (size_t)iblk * 32 * 128;
    #pragma unroll
    for (int it = 0; it < 16; ++it) {
      int e = tid + it * NTHR;            // coalesced
      float w = Wsl[e];
      int j = e >> 7, s = e & 127;
      unsigned int h, l;
      bf16_split(w, h, l);
      int b0 = j << 8;
      s_whi [b0 + s]       = (unsigned short)h;
      s_whi [b0 + s + 128] = (unsigned short)h;
      s_wlo [b0 + s]       = (unsigned short)l;
      s_wlo [b0 + s + 128] = (unsigned short)l;
      int c0 = (s + 255) & 255;           // positions c with (c+1)%256 == s
      int c1 = (s + 127) & 255;           // ... == s+128
      s_whiS[b0 + c0] = (unsigned short)h;
      s_whiS[b0 + c1] = (unsigned short)h;
      s_wloS[b0 + c0] = (unsigned short)l;
      s_wloS[b0 + c1] = (unsigned short)l;
    }
  }

  v8f acc[2][4];
  #pragma unroll
  for (int mt = 0; mt < 2; ++mt)
    #pragma unroll
    for (int nt = 0; nt < 4; ++nt)
      acc[mt][nt] = (v8f){0.f, 0.f, 0.f, 0.f, 0.f, 0.f, 0.f, 0.f};

  float bv[4];
  #pragma unroll
  for (int nt = 0; nt < 4; ++nt)
    bv[nt] = bias[n0 + wn * 64 + nt * 16 + ln];

  // A-tile global load assignment: 8 threads x float4 per 32-float row
  const int r_base = tid >> 3;   // 0..31
  const int c4     = tid & 7;    // 0..7

  float4 xreg[4];
  float4 dreg;
  {
    dreg = *(const float4*)&dvec[c4 * 4];
    #pragma unroll
    for (int rr = 0; rr < 4; ++rr) {
      int row = r_base + rr * 32;
      xreg[rr] = *(const float4*)&x[(size_t)(m0 + row) * DIN + c4 * 4];
    }
  }

  const unsigned int* whi32  = (const unsigned int*)s_whi;
  const unsigned int* whiS32 = (const unsigned int*)s_whiS;
  const unsigned int* wlo32  = (const unsigned int*)s_wlo;
  const unsigned int* wloS32 = (const unsigned int*)s_wloS;

  for (int kk = 0; kk < DIN / TILE_K; ++kk) {
    // ---- store staged x regs -> LDS as bf16 hi/lo pairs ----
    #pragma unroll
    for (int rr = 0; rr < 4; ++rr) {
      int row = r_base + rr * 32;
      float4 v = xreg[rr];
      v.x *= dreg.x; v.y *= dreg.y; v.z *= dreg.z; v.w *= dreg.w;
      unsigned int h0, l0, h1, l1, h2, l2, h3, l3;
      bf16_split(v.x, h0, l0); bf16_split(v.y, h1, l1);
      bf16_split(v.z, h2, l2); bf16_split(v.w, h3, l3);
      int idx = row * 18 + c4 * 2;
      *(uint2*)&s_xhi[idx] = make_uint2(h0 | (h1 << 16), h2 | (h3 << 16));
      *(uint2*)&s_xlo[idx] = make_uint2(l0 | (l1 << 16), l2 | (l3 << 16));
    }
    __syncthreads();   // first iteration also covers the W staging

    // ---- prefetch next x tile into registers (hides global latency) ----
    if (kk + 1 < DIN / TILE_K) {
      int k0n = (kk + 1) * TILE_K;
      dreg = *(const float4*)&dvec[k0n + c4 * 4];
      #pragma unroll
      for (int rr = 0; rr < 4; ++rr) {
        int row = r_base + rr * 32;
        xreg[rr] = *(const float4*)&x[(size_t)(m0 + row) * DIN + k0n + c4 * 4];
      }
    }

    // ---- build fragments from LDS ----
    const int j    = kk >> 2;           // input circulant block
    const int u0   = (kk & 3) << 5;     // K offset within block
    const int wrow = j << 7;            // row offset in u32 units

    BFrag ah[2], al[2], bh[4], bl[4];

    #pragma unroll
    for (int mt = 0; mt < 2; ++mt) {
      int row = wm * 32 + mt * 16 + ln;
      const unsigned int* ph = &s_xhi[row * 18 + 8 * lg];
      const unsigned int* pl = &s_xlo[row * 18 + 8 * lg];
      #pragma unroll
      for (int v = 0; v < 8; ++v) {
        int off = v;                             // A layout: K=2v+p+16*lg
        ah[mt].u[v] = ph[off];
        al[mt].u[v] = pl[off];
      }
    }

    #pragma unroll
    for (int nt = 0; nt < 4; ++nt) {
      int t    = wn * 64 + nt * 16 + ln;         // output column within block
      int base = 128 + u0 + (lg << 4) - t;       // index into dup'd w row, in [1,255]
      int c2   = base >> 1;
      const unsigned int* ph = ((base & 1) ? whiS32 : whi32) + wrow + c2;
      const unsigned int* pl = ((base & 1) ? wloS32 : wlo32) + wrow + c2;
      #pragma unroll
      for (int v = 0; v < 8; ++v) {              // B layout: K=2v+p+16*lg
        bh[nt].u[v] = ph[v];
        bl[nt].u[v] = pl[v];
      }
    }

    // ---- WMMA: hi*hi + lo*hi + hi*lo (f32 accumulate) ----
    #pragma unroll
    for (int mt = 0; mt < 2; ++mt) {
      #pragma unroll
      for (int nt = 0; nt < 4; ++nt) {
        acc[mt][nt] = __builtin_amdgcn_wmma_f32_16x16x32_bf16(
            false, ah[mt].v, false, bh[nt].v, (short)0, acc[mt][nt], false, false);
        acc[mt][nt] = __builtin_amdgcn_wmma_f32_16x16x32_bf16(
            false, al[mt].v, false, bh[nt].v, (short)0, acc[mt][nt], false, false);
        acc[mt][nt] = __builtin_amdgcn_wmma_f32_16x16x32_bf16(
            false, ah[mt].v, false, bl[nt].v, (short)0, acc[mt][nt], false, false);
      }
    }
    __syncthreads();   // LDS reads done before next iteration's stores
  }

  // ---- epilogue: C/D layout M=r+8*lg, N=ln ----
  #pragma unroll
  for (int mt = 0; mt < 2; ++mt) {
    #pragma unroll
    for (int nt = 0; nt < 4; ++nt) {
      int col  = n0 + wn * 64 + nt * 16 + ln;
      int rowb = m0 + wm * 32 + mt * 16 + 8 * lg;
      #pragma unroll
      for (int r = 0; r < 8; ++r) {
        y[(size_t)(rowb + r) * DOUT + col] = acc[mt][nt][r] + bv[nt];
      }
    }
  }
}

extern "C" void kernel_launch(void* const* d_in, const int* in_sizes, int n_in,
                              void* d_out, int out_size, void* d_ws, size_t ws_size,
                              hipStream_t stream) {
  const float* x    = (const float*)d_in[0];
  const float* W    = (const float*)d_in[1];
  const float* dvec = (const float*)d_in[2];
  const float* bias = (const float*)d_in[3];
  float* y = (float*)d_out;

  const int batch = in_sizes[0] / DIN;          // 8192
  dim3 grid(batch / TILE_M, DOUT / CB);         // (64, 32)
  dim3 block(NTHR);
  bc_wmma_kernel<<<grid, block, 0, stream>>>(x, W, dvec, bias, y);
}
